// SpikingLinear_12567074308280
// MI455X (gfx1250) — matrix operand order repaired and checked by reading, not compiled, not run on hardware
//
#include <hip/hip_runtime.h>

typedef __attribute__((ext_vector_type(16))) _Float16 v16h;
typedef __attribute__((ext_vector_type(8)))  _Float16 v8h;
typedef __attribute__((ext_vector_type(8)))  float    v8f;

namespace {
constexpr int kB    = 64;
constexpr int kIn   = 1024;
constexpr int kOut  = 1024;
constexpr int kS    = 100;
constexpr int kSE   = kS - 1;          // 99 projected steps
constexpr int kRows = kSE * kB;        // 6336 GEMM rows, row = t*64 + b
}

// ---------------------------------------------------------------------------
// Kernel 1: x[B][IN][S] f32  ->  xT[row=t*B+b][i] f16 (GEMM A matrix, row-major)
// ---------------------------------------------------------------------------
__global__ void cvt_x(const float* __restrict__ x, _Float16* __restrict__ xT) {
  int idx = blockIdx.x * blockDim.x + threadIdx.x;   // over kRows*kIn, i fastest
  int i   = idx & (kIn - 1);
  int row = idx >> 10;                               // kIn == 1024
  int b   = row & (kB - 1);
  int t   = row >> 6;                                // kB == 64
  xT[idx] = (_Float16)x[(size_t)(b * kIn + i) * kS + t];
}

// ---------------------------------------------------------------------------
// Kernel 2: W f32 -> f16
// ---------------------------------------------------------------------------
__global__ void cvt_w(const float* __restrict__ W, _Float16* __restrict__ Wh) {
  int idx = blockIdx.x * blockDim.x + threadIdx.x;
  Wh[idx] = (_Float16)W[idx];
}

// ---------------------------------------------------------------------------
// Kernel 3: WMMA GEMM  proj[row][o] = sum_i xT[row][i] * Wh[o][i]
//   block = 128 threads = 4 waves; wave w owns a 64x32 output tile:
//   rows [blockIdx.y*64, +64), cols [(blockIdx.x*4+w)*32, +32).
//   A registers are reused across the two N-tiles (B over 4 M-tiles: 6 operand loads per 8 WMMAs).
// ---------------------------------------------------------------------------
__global__ __launch_bounds__(128) void gemm_wmma(const _Float16* __restrict__ xT,
                                                 const _Float16* __restrict__ Wh,
                                                 float* __restrict__ proj) {
  const int lane  = threadIdx.x & 31;
  const int wave  = threadIdx.x >> 5;
  const int half  = lane >> 4;          // 0: lanes 0-15, 1: lanes 16-31
  const int l     = lane & 15;
  const int nBase = (blockIdx.x * 4 + wave) * 32;
  const int mBase = blockIdx.y * 64;

  v8f acc00 = {}, acc01 = {}, acc10 = {}, acc11 = {};
  v8f acc20 = {}, acc21 = {}, acc30 = {}, acc31 = {};

  // A rows: lane L and L+16 both map to M = l within a 16-row tile.
  const _Float16* a0 = xT + (size_t)(mBase +  0 + l) * kIn;
  const _Float16* a1 = xT + (size_t)(mBase + 16 + l) * kIn;
  const _Float16* a2 = xT + (size_t)(mBase + 32 + l) * kIn;
  const _Float16* a3 = xT + (size_t)(mBase + 48 + l) * kIn;
  // B: lane column N; low half-wave K 0-15, high half-wave K 16-31.
  const _Float16* bp0 = Wh + (size_t)(nBase + l) * kIn + half * 16;
  const _Float16* bp1 = bp0 + (size_t)16 * kIn;
  // A per-lane K chunks: elems 0-7 -> K0 + half*16, elems 8-15 -> K0+8 + half*16
  const int aoff0 = half * 16;
  const int aoff1 = 8 + half * 16;

  union AV { v16h v; v8h h[2]; };

  #pragma unroll 2
  for (int k0 = 0; k0 < kIn; k0 += 32) {
    v16h b0 = *(const v16h*)(bp0 + k0);
    v16h b1 = *(const v16h*)(bp1 + k0);

    AV a;
    a.h[0] = *(const v8h*)(a0 + k0 + aoff0);
    a.h[1] = *(const v8h*)(a0 + k0 + aoff1);
    acc00 = __builtin_amdgcn_wmma_f32_16x16x32_f16(false, a.v, false, b0,
                                                   (short)0, acc00, false, false);
    acc01 = __builtin_amdgcn_wmma_f32_16x16x32_f16(false, a.v, false, b1,
                                                   (short)0, acc01, false, false);
    a.h[0] = *(const v8h*)(a1 + k0 + aoff0);
    a.h[1] = *(const v8h*)(a1 + k0 + aoff1);
    acc10 = __builtin_amdgcn_wmma_f32_16x16x32_f16(false, a.v, false, b0,
                                                   (short)0, acc10, false, false);
    acc11 = __builtin_amdgcn_wmma_f32_16x16x32_f16(false, a.v, false, b1,
                                                   (short)0, acc11, false, false);
    a.h[0] = *(const v8h*)(a2 + k0 + aoff0);
    a.h[1] = *(const v8h*)(a2 + k0 + aoff1);
    acc20 = __builtin_amdgcn_wmma_f32_16x16x32_f16(false, a.v, false, b0,
                                                   (short)0, acc20, false, false);
    acc21 = __builtin_amdgcn_wmma_f32_16x16x32_f16(false, a.v, false, b1,
                                                   (short)0, acc21, false, false);
    a.h[0] = *(const v8h*)(a3 + k0 + aoff0);
    a.h[1] = *(const v8h*)(a3 + k0 + aoff1);
    acc30 = __builtin_amdgcn_wmma_f32_16x16x32_f16(false, a.v, false, b0,
                                                   (short)0, acc30, false, false);
    acc31 = __builtin_amdgcn_wmma_f32_16x16x32_f16(false, a.v, false, b1,
                                                   (short)0, acc31, false, false);
  }

  // D layout: VGPR r -> M = half*8 + r (per 16-row tile), N = l (per 16-col tile).
  float* p0 = proj + (size_t)(mBase + half * 8) * kOut + nBase + l;
  #pragma unroll
  for (int r = 0; r < 8; ++r) {
    p0[(size_t)(r)      * kOut]      = acc00[r];
    p0[(size_t)(r)      * kOut + 16] = acc01[r];
    p0[(size_t)(16 + r) * kOut]      = acc10[r];
    p0[(size_t)(16 + r) * kOut + 16] = acc11[r];
    p0[(size_t)(32 + r) * kOut]      = acc20[r];
    p0[(size_t)(32 + r) * kOut + 16] = acc21[r];
    p0[(size_t)(48 + r) * kOut]      = acc30[r];
    p0[(size_t)(48 + r) * kOut + 16] = acc31[r];
  }
}

// ---------------------------------------------------------------------------
// Kernel 4: sequential LIF scan. One thread per (b,o). proj reads coalesced.
// ---------------------------------------------------------------------------
__global__ void lif_scan(const float* __restrict__ proj, float* __restrict__ out) {
  const int tid = blockIdx.x * blockDim.x + threadIdx.x;   // tid = b*kOut + o
  const float a_m = 1.0f - 1.0f / 20.0f;   // 0.95
  const float b_m = 1.0f / 20.0f;          // 0.05
  const float a_s = 1.0f - 1.0f / 5.0f;    // 0.8

  float V = 0.0f, I = 0.0f;
  float* orow = out + (size_t)tid * kS;
  orow[0] = 0.0f;                          // step 0 is zeros
  for (int t = 0; t < kSE; ++t) {
    float p  = proj[(size_t)t * (kB * kOut) + tid];
    float Vn = a_m * V + b_m * I;
    float In = a_s * I + p;
    float s  = (Vn > 1.0f) ? 1.0f : 0.0f;
    V = (1.0f - s) * Vn;
    I = In;
    orow[t + 1] = s;
  }
}

// ---------------------------------------------------------------------------
extern "C" void kernel_launch(void* const* d_in, const int* in_sizes, int n_in,
                              void* d_out, int out_size, void* d_ws, size_t ws_size,
                              hipStream_t stream) {
  const float* x = (const float*)d_in[0];   // [B, IN, S] f32
  const float* W = (const float*)d_in[1];   // [OUT, IN] f32
  float* out = (float*)d_out;               // [B, OUT, S] f32

  char* ws = (char*)d_ws;
  size_t off = 0;
  _Float16* xT = (_Float16*)(ws + off);
  off += (size_t)kRows * kIn * sizeof(_Float16);            // 12,976,128 B
  off = (off + 255) & ~(size_t)255;
  _Float16* Wh = (_Float16*)(ws + off);
  off += (size_t)kOut * kIn * sizeof(_Float16);             //  2,097,152 B
  off = (off + 255) & ~(size_t)255;
  float* proj = (float*)(ws + off);                         // 25,952,256 B

  cvt_x<<<(kRows * kIn) / 256, 256, 0, stream>>>(x, xT);
  cvt_w<<<(kOut * kIn) / 256, 256, 0, stream>>>(W, Wh);
  gemm_wmma<<<dim3(kOut / 128, kRows / 64), 128, 0, stream>>>(xT, Wh, proj);
  lif_scan<<<(kB * kOut) / 256, 256, 0, stream>>>(proj, out);
}
